// AttentionLayer_65841848648249
// MI455X (gfx1250) — compile-verified, matrix operand order checked by reading, not run on hardware
//
#include <hip/hip_runtime.h>
#include <math.h>

typedef __bf16 bf16_t;
typedef __attribute__((ext_vector_type(16))) __bf16 v16bf;
typedef __attribute__((ext_vector_type(8)))  float  v8f;

#define B_   32
#define T_   8192
#define D_   256
#define CTX_ 128
#define ROWS (B_ * T_)          // 262144 rows total
#define TILES (ROWS / 16)       // 16384 16-row tiles
#define K1_BLOCKS (TILES / 8)   // 8 waves per block, 1 tile per wave

// gfx1250 has a native transcendental V_TANH_F32; use it if clang exposes it.
#if defined(__has_builtin)
# if __has_builtin(__builtin_amdgcn_tanhf)
#  define FAST_TANH(x) __builtin_amdgcn_tanhf(x)
# elif __has_builtin(__builtin_amdgcn_tanh_f32)
#  define FAST_TANH(x) __builtin_amdgcn_tanh_f32(x)
# endif
#endif
#ifndef FAST_TANH
# define FAST_TANH(x) tanhf(x)
#endif

// Build an A fragment (16 bf16 per lane) per the ISA 16-bit A layout:
// lane holds row lane%16; element j -> K = kb + j + (j>=8 ? 8 : 0), where
// kb = ks*32 + (lane>>4)*8.  Two contiguous 32B chunks -> 2x b128 loads + cvt.
__device__ __forceinline__ v16bf load_a_frag(const float* __restrict__ xrow,
                                             int kb) {
    v16bf a;
#pragma unroll
    for (int j = 0; j < 8; ++j) {
        a[j]     = (bf16_t)xrow[kb + j];
        a[j + 8] = (bf16_t)xrow[kb + 16 + j];
    }
    return a;
}

// -------------------------------------------------------------------------
// K1: scores[row] = tanh(X[row,:] @ W) . u   via v_wmma_f32_16x16x32_bf16
// -------------------------------------------------------------------------
__global__ __launch_bounds__(256) void k1_scores_wmma(
    const float* __restrict__ X, const float* __restrict__ W,
    const float* __restrict__ u, float* __restrict__ scores)
{
    // Pre-swizzled B fragments: [nt=CTX tile][ks=K step][lane] -> 16 bf16
    __shared__ v16bf sWf[8 * 8 * 32];   // 64 KB

    const int tid   = threadIdx.x;
    const int lane  = tid & 31;
    const int wave  = tid >> 5;
    const int col   = lane & 15;   // N column within 16-wide tile / A row id
    const int khalf = lane >> 4;   // K half selector per ISA 16-bit layout

    // Cooperative fill of W fragments in the exact WMMA B layout.
    bf16_t* sw = (bf16_t*)sWf;
    for (int idx = tid; idx < 8 * 8 * 32 * 16; idx += 256) {
        const int j  = idx & 15;
        const int ln = (idx >> 4) & 31;
        const int ks = (idx >> 9) & 7;
        const int nt = idx >> 12;
        const int k  = ks * 32 + (ln >> 4) * 8 + j + ((j >= 8) ? 8 : 0);
        const int c  = nt * 16 + (ln & 15);
        sw[idx] = (bf16_t)W[k * CTX_ + c];
    }
    __syncthreads();

    const long tile = (long)blockIdx.x * 8 + wave;
    const long row0 = tile * 16;                       // never straddles a batch
    const float* xrow = X + (row0 + col) * (long)D_;   // A: lane = row (col)

    v8f acc[8] = {};   // 8 N-tiles of f32 C/D accumulators

    // Ping-pong A fragments: every prefetch is a fresh load into its own
    // registers (no cross-iteration register copies -> no movs / hazard nops).
    v16bf a0 = load_a_frag(xrow, khalf * 8);           // ks = 0

#pragma unroll 1
    for (int ks = 0; ks < 8; ks += 2) {
        v16bf b[8];
#pragma unroll
        for (int nt = 0; nt < 8; ++nt) b[nt] = sWf[(nt * 8 + ks) * 32 + lane];

        v16bf a1 = load_a_frag(xrow, (ks + 1) * 32 + khalf * 8);

#pragma unroll
        for (int nt = 0; nt < 8; ++nt)
            acc[nt] = __builtin_amdgcn_wmma_f32_16x16x32_bf16(
                false, a0, false, b[nt], (short)0, acc[nt], false, false);

#pragma unroll
        for (int nt = 0; nt < 8; ++nt)
            b[nt] = sWf[(nt * 8 + ks + 1) * 32 + lane];

        if (ks < 6) a0 = load_a_frag(xrow, (ks + 2) * 32 + khalf * 8);

#pragma unroll
        for (int nt = 0; nt < 8; ++nt)
            acc[nt] = __builtin_amdgcn_wmma_f32_16x16x32_bf16(
                false, a1, false, b[nt], (short)0, acc[nt], false, false);
    }

    float uv[8];
#pragma unroll
    for (int nt = 0; nt < 8; ++nt) uv[nt] = u[nt * 16 + col];

    // C/D layout: VGPR r holds row (khalf*8 + r), lane%16 = column.
    // score[m] = sum_c tanh(val[m,c]) * u[c]  -> 16-lane xor-shuffle reduce.
#pragma unroll
    for (int r = 0; r < 8; ++r) {
        float p = 0.f;
#pragma unroll
        for (int nt = 0; nt < 8; ++nt) p += FAST_TANH(acc[nt][r]) * uv[nt];
        p += __shfl_xor(p, 1, 16);
        p += __shfl_xor(p, 2, 16);
        p += __shfl_xor(p, 4, 16);
        p += __shfl_xor(p, 8, 16);
        if (col == 0) scores[row0 + khalf * 8 + r] = p;
    }
}

// -------------------------------------------------------------------------
// K2: per-batch softmax over T (in place) + zero the output row
// -------------------------------------------------------------------------
__global__ __launch_bounds__(1024) void k2_softmax(
    float* __restrict__ scores, float* __restrict__ out)
{
    const int b = blockIdx.x;
    float* s = scores + (long)b * T_;

    __shared__ float red[32];
    __shared__ float bcast;
    const int tid  = threadIdx.x;
    const int lane = tid & 31;
    const int wv   = tid >> 5;

    float m = -3.402823466e38f;
    for (int t = tid; t < T_; t += 1024) m = fmaxf(m, s[t]);
    for (int off = 16; off; off >>= 1) m = fmaxf(m, __shfl_xor(m, off, 32));
    if (lane == 0) red[wv] = m;
    __syncthreads();
    if (wv == 0) {
        float v = red[lane];
        for (int off = 16; off; off >>= 1) v = fmaxf(v, __shfl_xor(v, off, 32));
        if (lane == 0) bcast = v;
    }
    __syncthreads();
    m = bcast;

    float sum = 0.f;
    for (int t = tid; t < T_; t += 1024) sum += expf(s[t] - m);
    for (int off = 16; off; off >>= 1) sum += __shfl_xor(sum, off, 32);
    if (lane == 0) red[wv] = sum;
    __syncthreads();
    if (wv == 0) {
        float v = red[lane];
        for (int off = 16; off; off >>= 1) v += __shfl_xor(v, off, 32);
        if (lane == 0) bcast = v;
    }
    __syncthreads();
    const float inv = 1.f / bcast;

    for (int t = tid; t < T_; t += 1024) s[t] = expf(s[t] - m) * inv;

    if (tid < D_) out[b * D_ + tid] = 0.f;   // harness poisons d_out
}

// -------------------------------------------------------------------------
// K3: out[b,d] = sum_t att[b,t] * X[b,t,d]   (memory-bound pass over X)
// -------------------------------------------------------------------------
__global__ __launch_bounds__(256) void k3_wsum(
    const float* __restrict__ X, const float* __restrict__ att,
    float* __restrict__ out)
{
    const int b     = blockIdx.y;
    const int chunk = blockIdx.x;          // T_/256 chunks of 256 timesteps
    const int d     = threadIdx.x;         // coalesced along D

    const float* xb = X   + ((long)b * T_ + (long)chunk * 256) * D_;
    const float* ab = att + (long)b * T_ + (long)chunk * 256;   // block-uniform

    float acc = 0.f;
#pragma unroll 4
    for (int t = 0; t < 256; ++t)
        acc = fmaf(ab[t], xb[(long)t * D_ + d], acc);

    atomicAdd(&out[b * D_ + d], acc);
}

// -------------------------------------------------------------------------
extern "C" void kernel_launch(void* const* d_in, const int* in_sizes, int n_in,
                              void* d_out, int out_size, void* d_ws, size_t ws_size,
                              hipStream_t stream) {
    const float* X = (const float*)d_in[0];   // (B,T,D) fp32
    const float* W = (const float*)d_in[1];   // (D,CTX) fp32
    const float* u = (const float*)d_in[2];   // (CTX,1) fp32
    float* out     = (float*)d_out;           // (B,D) fp32
    float* scores  = (float*)d_ws;            // (B,T) fp32 scratch, 1 MB

    k1_scores_wmma<<<K1_BLOCKS, 256, 0, stream>>>(X, W, u, scores);
    k2_softmax<<<B_, 1024, 0, stream>>>(scores, out);
    dim3 g3(T_ / 256, B_);
    k3_wsum<<<g3, 256, 0, stream>>>(X, scores, out);
}
